// HybridDistillationLoss_2456721293349
// MI455X (gfx1250) — matrix operand order repaired and checked by reading, not compiled, not run on hardware
//
#include <hip/hip_runtime.h>
#include <hip/hip_bf16.h>
#include <math.h>

// ---------------------------------------------------------------------------
// HybridDistillationLoss for MI455X (gfx1250, wave32, WMMA)
//
// Dominant cost: sn @ sn^T (4096x4096x1024, 34.4 GFLOP) feeding a global
// off-diagonal logsumexp. Computed on V_WMMA_F32_16X16X32_BF16 using a
// bf16 hi/lo split (3 WMMAs per tile-step => near-fp32 accuracy at ~2.7x
// the fp32-WMMA rate), upper-triangle only (symmetry halves the work).
// ---------------------------------------------------------------------------

typedef __attribute__((ext_vector_type(16))) __bf16 v16bf;
typedef __attribute__((ext_vector_type(8)))  float  v8f;

#define NB 4096
#define ND 1024
#define NTILE 128                       // 4096 / 32 wave-tiles per dim
#define NWTILES (NTILE * (NTILE + 1) / 2)  // 8256 upper-tri wave-tiles

// ---- workspace layout (bytes) ---------------------------------------------
static const size_t OFF_SN  = 0;                       // sn  f32  16 MB
static const size_t OFF_TN  = OFF_SN  + (size_t)NB*ND*4;   // tn  f32  16 MB
static const size_t OFF_AHI = OFF_TN  + (size_t)NB*ND*4;   // sn  hi bf16 row-major 8 MB
static const size_t OFF_ALO = OFF_AHI + (size_t)NB*ND*2;   // sn  lo bf16 row-major 8 MB
static const size_t OFF_BHI = OFF_ALO + (size_t)NB*ND*2;   // snT hi bf16 (K-major)  8 MB
static const size_t OFF_BLO = OFF_BHI + (size_t)NB*ND*2;   // snT lo bf16 (K-major)  8 MB
static const size_t OFF_S16 = OFF_BLO + (size_t)NB*ND*2;   // S rows 16x4096 f32
static const size_t OFF_T16 = OFF_S16 + (size_t)16*NB*4;   // T rows 16x4096 f32
static const size_t OFF_RD  = OFF_T16 + (size_t)16*NB*4;   // per-row align dots 4096 f32
static const size_t OFF_PM  = OFF_RD  + (size_t)NB*4;      // per-wave-tile max
static const size_t OFF_PS  = OFF_PM  + (size_t)NWTILES*4; // per-wave-tile sumexp
static const size_t OFF_SC  = OFF_PS  + (size_t)NWTILES*4; // scalars [align_sum, unif]

__device__ __forceinline__ unsigned short f2bf(float f) {
  unsigned u = __float_as_uint(f);
  unsigned r = u + 0x7FFFu + ((u >> 16) & 1u);   // round-to-nearest-even
  return (unsigned short)(r >> 16);
}
__device__ __forceinline__ float bf2f(unsigned short h) {
  return __uint_as_float(((unsigned)h) << 16);
}

// A-fragment: 16-bit A 16x32 layout: lane<16 -> M=lane, K = {k..k+7, k+16..k+23}
// (two contiguous 16B chunks, 32B apart). lanes 16..31 use k+8 base.
__device__ __forceinline__ v16bf load_a(const unsigned short* __restrict__ p,
                                        int row, int k) {
  const unsigned char* q = (const unsigned char*)(p + (size_t)row * ND + k);
  struct { uint4 x, y; } tmp;
  tmp.x = *(const uint4*)q;
  tmp.y = *(const uint4*)(q + 32);
  v16bf f; __builtin_memcpy(&f, &tmp, 32);
  return f;
}
// Row loader: N = col..col+15 at one K (unused; the GEMM loads B via load_a).
// From the K-major (transposed) copy this is 32 contiguous bytes.
__device__ __forceinline__ v16bf load_b(const unsigned short* __restrict__ p,
                                        int k, int col) {
  const unsigned char* q = (const unsigned char*)(p + (size_t)k * NB + col);
  struct { uint4 x, y; } tmp;
  tmp.x = *(const uint4*)q;
  tmp.y = *(const uint4*)(q + 16);
  v16bf f; __builtin_memcpy(&f, &tmp, 32);
  return f;
}

// ---------------------------------------------------------------------------
// Kernel A: per-row L2 normalize of student/teacher, bf16 hi/lo split of sn
// (row-major + K-major copies), and per-row alignment dot.
// One block (256 threads) per row; 4 elements per thread.
// ---------------------------------------------------------------------------
__global__ __launch_bounds__(256) void normalize_split_kernel(
    const float* __restrict__ se, const float* __restrict__ te,
    float* __restrict__ sn, float* __restrict__ tn,
    unsigned short* __restrict__ ahi, unsigned short* __restrict__ alo,
    unsigned short* __restrict__ bhi, unsigned short* __restrict__ blo,
    float* __restrict__ row_dot) {
  __shared__ float red[256];
  const int row = blockIdx.x;
  const int t = threadIdx.x;
  const float* srow = se + (size_t)row * ND;
  const float* trow = te + (size_t)row * ND;

  float s[4], u[4];
#pragma unroll
  for (int i = 0; i < 4; ++i) { s[i] = srow[t + 256 * i]; u[i] = trow[t + 256 * i]; }

  float ss = s[0]*s[0] + s[1]*s[1] + s[2]*s[2] + s[3]*s[3];
  float tt = u[0]*u[0] + u[1]*u[1] + u[2]*u[2] + u[3]*u[3];

  red[t] = ss; __syncthreads();
  for (int o = 128; o > 0; o >>= 1) { if (t < o) red[t] += red[t + o]; __syncthreads(); }
  const float sinv = 1.0f / fmaxf(sqrtf(red[0]), 1e-12f);
  __syncthreads();
  red[t] = tt; __syncthreads();
  for (int o = 128; o > 0; o >>= 1) { if (t < o) red[t] += red[t + o]; __syncthreads(); }
  const float tinv = 1.0f / fmaxf(sqrtf(red[0]), 1e-12f);
  __syncthreads();

  float dp = 0.0f;
#pragma unroll
  for (int i = 0; i < 4; ++i) {
    const int k = t + 256 * i;
    const float sv = s[i] * sinv;
    const float tv = u[i] * tinv;
    sn[(size_t)row * ND + k] = sv;
    tn[(size_t)row * ND + k] = tv;
    const unsigned short hi = f2bf(sv);
    const unsigned short lo = f2bf(sv - bf2f(hi));
    ahi[(size_t)row * ND + k] = hi;           // row-major (A fragments)
    alo[(size_t)row * ND + k] = lo;
    bhi[(size_t)k * NB + row] = hi;           // K-major (B fragments)
    blo[(size_t)k * NB + row] = lo;
    dp += sv * tv;
  }
  red[t] = dp; __syncthreads();
  for (int o = 128; o > 0; o >>= 1) { if (t < o) red[t] += red[t + o]; __syncthreads(); }
  if (t == 0) row_dot[row] = red[0];
}

// ---------------------------------------------------------------------------
// Kernel B: upper-triangle Gram GEMM with online max/sum-exp.
// One wave per 32x32 output tile (2x2 WMMA sub-tiles), bf16 hi/lo split:
// acc += hiA*hiB + hiA*loB + loA*hiB  (12 v_wmma per K-step of 32).
// ---------------------------------------------------------------------------
__global__ __launch_bounds__(256) void uniformity_gemm_kernel(
    const unsigned short* __restrict__ ahi, const unsigned short* __restrict__ alo,
    const unsigned short* __restrict__ bhi, const unsigned short* __restrict__ blo,
    float* __restrict__ pmax, float* __restrict__ psum) {
  const int wave = (int)((blockIdx.x * blockDim.x + threadIdx.x) >> 5);
  const int lane = threadIdx.x & 31;
  if (wave >= NWTILES) return;

  // upper-triangle wave-tile index -> (ti, tj), ti <= tj
  int ti = 0, rem = wave;
  while (rem >= NTILE - ti) { rem -= NTILE - ti; ++ti; }
  const int tj = ti + rem;
  const int rowBase = ti * 32, colBase = tj * 32;

  const v8f zero = {0.f, 0.f, 0.f, 0.f, 0.f, 0.f, 0.f, 0.f};
  v8f acc[2][2] = {{zero, zero}, {zero, zero}};

  const int mrow0 = rowBase + (lane & 15);
  const int mrow1 = mrow0 + 16;
  const int kSel = (lane < 16) ? 0 : 8;

  for (int k0 = 0; k0 < ND; k0 += 32) {
    const v16bf a0h = load_a(ahi, mrow0, k0 + kSel);
    const v16bf a0l = load_a(alo, mrow0, k0 + kSel);
    const v16bf a1h = load_a(ahi, mrow1, k0 + kSel);
    const v16bf a1l = load_a(alo, mrow1, k0 + kSel);
    const v16bf b0h = load_a(ahi, colBase + (lane & 15), k0 + kSel);
    const v16bf b0l = load_a(alo, colBase + (lane & 15), k0 + kSel);
    const v16bf b1h = load_a(ahi, colBase + 16 + (lane & 15), k0 + kSel);
    const v16bf b1l = load_a(alo, colBase + 16 + (lane & 15), k0 + kSel);

    acc[0][0] = __builtin_amdgcn_wmma_f32_16x16x32_bf16(false, a0h, false, b0h, (short)0, acc[0][0], false, false);
    acc[0][0] = __builtin_amdgcn_wmma_f32_16x16x32_bf16(false, a0h, false, b0l, (short)0, acc[0][0], false, false);
    acc[0][0] = __builtin_amdgcn_wmma_f32_16x16x32_bf16(false, a0l, false, b0h, (short)0, acc[0][0], false, false);

    acc[0][1] = __builtin_amdgcn_wmma_f32_16x16x32_bf16(false, a0h, false, b1h, (short)0, acc[0][1], false, false);
    acc[0][1] = __builtin_amdgcn_wmma_f32_16x16x32_bf16(false, a0h, false, b1l, (short)0, acc[0][1], false, false);
    acc[0][1] = __builtin_amdgcn_wmma_f32_16x16x32_bf16(false, a0l, false, b1h, (short)0, acc[0][1], false, false);

    acc[1][0] = __builtin_amdgcn_wmma_f32_16x16x32_bf16(false, a1h, false, b0h, (short)0, acc[1][0], false, false);
    acc[1][0] = __builtin_amdgcn_wmma_f32_16x16x32_bf16(false, a1h, false, b0l, (short)0, acc[1][0], false, false);
    acc[1][0] = __builtin_amdgcn_wmma_f32_16x16x32_bf16(false, a1l, false, b0h, (short)0, acc[1][0], false, false);

    acc[1][1] = __builtin_amdgcn_wmma_f32_16x16x32_bf16(false, a1h, false, b1h, (short)0, acc[1][1], false, false);
    acc[1][1] = __builtin_amdgcn_wmma_f32_16x16x32_bf16(false, a1h, false, b1l, (short)0, acc[1][1], false, false);
    acc[1][1] = __builtin_amdgcn_wmma_f32_16x16x32_bf16(false, a1l, false, b1h, (short)0, acc[1][1], false, false);
  }

  // C/D layout: lane<16 -> M=v, N=lane ; lane>=16 -> M=v+8, N=lane-16
  const bool diag = (ti == tj);
  const int mOff = (lane >> 4) << 3;   // +8 for upper half-wave
  const int nIdx = lane & 15;

  float lmax = -INFINITY;
#pragma unroll
  for (int m = 0; m < 2; ++m)
#pragma unroll
    for (int n = 0; n < 2; ++n)
#pragma unroll
      for (int v = 0; v < 8; ++v) {
        const int rg = rowBase + m * 16 + v + mOff;
        const int cg = colBase + n * 16 + nIdx;
        if (diag && cg <= rg) continue;        // strict upper triangle only
        lmax = fmaxf(lmax, acc[m][n][v] * 10.0f);   // /TEMP
      }
#pragma unroll
  for (int o = 16; o > 0; o >>= 1) lmax = fmaxf(lmax, __shfl_xor(lmax, o, 32));

  float lsum = 0.0f;
#pragma unroll
  for (int m = 0; m < 2; ++m)
#pragma unroll
    for (int n = 0; n < 2; ++n)
#pragma unroll
      for (int v = 0; v < 8; ++v) {
        const int rg = rowBase + m * 16 + v + mOff;
        const int cg = colBase + n * 16 + nIdx;
        if (diag && cg <= rg) continue;
        lsum += expf(acc[m][n][v] * 10.0f - lmax);
      }
#pragma unroll
  for (int o = 16; o > 0; o >>= 1) lsum += __shfl_xor(lsum, o, 32);

  if (lane == 0) { pmax[wave] = lmax; psum[wave] = lsum; }
}

// ---------------------------------------------------------------------------
// Kernel D: first-16-row Gram rows in fp32 (S16 = sn[:16]@sn^T, T16 likewise).
// One wave per output element, lanes parallel over K (coalesced).
// ---------------------------------------------------------------------------
__global__ __launch_bounds__(256) void small_rows_kernel(
    const float* __restrict__ sn, const float* __restrict__ tn,
    float* __restrict__ S16, float* __restrict__ T16) {
  const int wave = (int)((blockIdx.x * blockDim.x + threadIdx.x) >> 5);
  const int lane = threadIdx.x & 31;
  if (wave >= 2 * 16 * NB) return;
  const int sel = wave >> 16;          // 0: student, 1: teacher
  const int i   = (wave >> 12) & 15;
  const int j   = wave & (NB - 1);
  const float* base = sel ? tn : sn;
  const float* ri = base + (size_t)i * ND;
  const float* rj = base + (size_t)j * ND;
  float acc = 0.0f;
  for (int k = lane; k < ND; k += 32) acc += ri[k] * rj[k];
#pragma unroll
  for (int o = 16; o > 0; o >>= 1) acc += __shfl_xor(acc, o, 32);
  if (lane == 0) (sel ? T16 : S16)[i * NB + j] = acc;
}

// ---------------------------------------------------------------------------
// Kernel C: deterministic reductions — alignment sum + global LSE merge.
// ---------------------------------------------------------------------------
__global__ __launch_bounds__(256) void reduce_kernel(
    const float* __restrict__ row_dot,
    const float* __restrict__ pmax, const float* __restrict__ psum,
    float* __restrict__ scal) {
  __shared__ float sm[256], ss[256];
  const int t = threadIdx.x;

  float a = 0.0f;
  for (int i = t; i < NB; i += 256) a += row_dot[i];
  sm[t] = a; __syncthreads();
  for (int o = 128; o > 0; o >>= 1) { if (t < o) sm[t] += sm[t + o]; __syncthreads(); }
  const float align_sum = sm[0];
  __syncthreads();

  float m = -INFINITY, s = 0.0f;
  for (int i = t; i < NWTILES; i += 256) {
    const float pm = pmax[i], ps = psum[i];
    if (!(ps > 0.0f)) continue;
    if (pm > m) { s = s * expf(m - pm) + ps; m = pm; }
    else        { s += ps * expf(pm - m); }
  }
  sm[t] = m; ss[t] = s; __syncthreads();
  for (int o = 128; o > 0; o >>= 1) {
    if (t < o) {
      const float m1 = sm[t], s1 = ss[t];
      const float m2 = sm[t + o], s2 = ss[t + o];
      const float M = fmaxf(m1, m2);
      float S = 0.0f;
      if (M > -INFINITY) S = s1 * expf(m1 - M) + s2 * expf(m2 - M);
      sm[t] = M; ss[t] = S;
    }
    __syncthreads();
  }
  if (t == 0) {
    scal[0] = align_sum;
    scal[1] = sm[0] + logf(2.0f * ss[0]);   // symmetry: off-diag = 2 * upper
  }
}

// ---------------------------------------------------------------------------
// Kernel E: triplet ranking logic (A=16 rows) + final weighted combine.
// ---------------------------------------------------------------------------
__global__ __launch_bounds__(32) void final_kernel(
    const float* __restrict__ S16, const float* __restrict__ T16,
    const float* __restrict__ scal, float* __restrict__ out) {
  __shared__ float vio[16];
  __shared__ int   cnt[16];
  const int t = threadIdx.x;
  if (t < 16) {
    const float* Trow = T16 + t * NB;
    const float* Srow = S16 + t * NB;
    int pos = 0; float best = -INFINITY;
    for (int j = 0; j < NB; ++j) {
      const float v = (j == t) ? -1.0f : Trow[j];
      if (v > best) { best = v; pos = j; }
    }
    int neg = 0; float bestd = -INFINITY;
    for (int j = 0; j < NB; ++j) {
      const float d = (j == t || j == pos) ? -2.0f : (Srow[j] - Trow[j]);
      if (d > bestd) { bestd = d; neg = j; }
    }
    const bool valid = (neg != t) && (neg != pos);
    const float viol = fmaxf(Srow[neg] - Srow[pos] + 0.1f, 0.0f);
    vio[t] = valid ? viol : 0.0f;
    cnt[t] = valid ? 1 : 0;
  }
  __syncthreads();
  if (t == 0) {
    float rsum = 0.0f; int n = 0;
    for (int i = 0; i < 16; ++i) { rsum += vio[i]; n += cnt[i]; }
    const float ranking = (n > 0) ? (rsum / (float)n) : 0.0f;
    const float align = 1.0f - scal[0] / (float)NB;
    const float unif = scal[1];
    out[0] = 1.0f * align + 0.1f * unif + 0.4f * ranking;
  }
}

// ---------------------------------------------------------------------------
extern "C" void kernel_launch(void* const* d_in, const int* in_sizes, int n_in,
                              void* d_out, int out_size, void* d_ws, size_t ws_size,
                              hipStream_t stream) {
  (void)in_sizes; (void)n_in; (void)out_size; (void)ws_size;
  const float* se = (const float*)d_in[0];
  const float* te = (const float*)d_in[1];
  char* ws = (char*)d_ws;

  float*          sn  = (float*)(ws + OFF_SN);
  float*          tn  = (float*)(ws + OFF_TN);
  unsigned short* ahi = (unsigned short*)(ws + OFF_AHI);
  unsigned short* alo = (unsigned short*)(ws + OFF_ALO);
  unsigned short* bhi = (unsigned short*)(ws + OFF_BHI);
  unsigned short* blo = (unsigned short*)(ws + OFF_BLO);
  float*          S16 = (float*)(ws + OFF_S16);
  float*          T16 = (float*)(ws + OFF_T16);
  float*          rd  = (float*)(ws + OFF_RD);
  float*          pm  = (float*)(ws + OFF_PM);
  float*          ps  = (float*)(ws + OFF_PS);
  float*          sc  = (float*)(ws + OFF_SC);
  float*          out = (float*)d_out;

  normalize_split_kernel<<<NB, 256, 0, stream>>>(se, te, sn, tn, ahi, alo, bhi, blo, rd);

  const int gemmBlocks = (NWTILES * 32 + 255) / 256;   // 1032
  uniformity_gemm_kernel<<<gemmBlocks, 256, 0, stream>>>(ahi, alo, bhi, blo, pm, ps);

  small_rows_kernel<<<(2 * 16 * NB * 32) / 256, 256, 0, stream>>>(sn, tn, S16, T16);

  reduce_kernel<<<1, 256, 0, stream>>>(rd, pm, ps, sc);

  final_kernel<<<1, 32, 0, stream>>>(S16, T16, sc, out);
}
